// MultiheadAttention_18348100288705
// MI455X (gfx1250) — matrix operand order repaired and checked by reading, not compiled, not run on hardware
//
#include <hip/hip_runtime.h>
#include <hip/hip_bf16.h>

// Problem constants (match reference)
#define D_MODEL 1024
#define N_HEAD  16
#define D_HEAD  64
#define NB      4
#define SQ      2048
#define MTOT    (NB * SQ)        // 8192 rows for the big GEMMs

typedef __bf16 bf16;
typedef __attribute__((ext_vector_type(16))) __bf16 v16bf;
typedef __attribute__((ext_vector_type(8)))  __bf16 v8bf;
typedef __attribute__((ext_vector_type(8)))  float  v8f;
typedef unsigned int v4u __attribute__((ext_vector_type(4)));
typedef int          v4i __attribute__((ext_vector_type(4)));
typedef int          v8i __attribute__((ext_vector_type(8)));

// ---------------------------------------------------------------------------
// WMMA fragment loaders (CDNA5 ISA 7.12.2 layouts, 16-bit data, K=32 tiles)
// A (16x32, MxK): lane L holds row M=L&15; elems 0..7 = K at 8*(L>>4),
//                 elems 8..15 = K at 16 + 8*(L>>4).
// B (32x16, KxN): lane L holds col N=L&15; elems 0..7 = K at 8*(L>>4),
//                 elems 8..15 = K at 16 + 8*(L>>4).
// Tiles live in LDS row-major, leading dim 32 (A rows = M, B rows = N).
// ---------------------------------------------------------------------------
__device__ __forceinline__ v16bf load_fragA(const bf16* t, int row0) {
    int lane = threadIdx.x & 31;
    int r  = row0 + (lane & 15);
    int ko = (lane >> 4) << 3;                 // 0 or 8
    const bf16* p = t + r * 32 + ko;
    v8bf lo = *(const v8bf*)(p);
    v8bf hi = *(const v8bf*)(p + 16);
    v16bf f;
#pragma unroll
    for (int i = 0; i < 8; ++i) { f[i] = lo[i]; f[i + 8] = hi[i]; }
    return f;
}

__device__ __forceinline__ v16bf load_fragB(const bf16* t, int col0) {
    int lane = threadIdx.x & 31;
    int n  = col0 + (lane & 15);
    int ko = (lane >> 4) << 3;                 // 0 or 8
    const bf16* p = t + n * 32 + ko;
    v8bf lo = *(const v8bf*)(p);
    v8bf hi = *(const v8bf*)(p + 16);
    v16bf f;
#pragma unroll
    for (int i = 0; i < 8; ++i) { f[i] = lo[i]; f[i + 8] = hi[i]; }
    return f;
}

#define WMMA_BF16(A, Bm, C) \
    __builtin_amdgcn_wmma_f32_16x16x32_bf16(false, (A), false, (Bm), (short)0, (C), false, false)

// ---------------------------------------------------------------------------
// Tensor Data Mover: 2D tile load (global, row-major, elem=bf16) -> LDS.
// D# built per ISA ch.8: group0 = {count|flags, lds_addr, gaddr lo, gaddr hi|type},
// group1 = {mask/datasize/pad, tensor dims, tile dims, dim0 stride}.
// ---------------------------------------------------------------------------
__device__ __forceinline__ unsigned lds_off_u32(const void* p) {
    return (unsigned)(unsigned long long)(__attribute__((address_space(3))) const char*)p;
}

__device__ __forceinline__ void tdm_load_2d(unsigned lds_addr, const void* gptr,
                                            unsigned tile_k, unsigned tile_rows,
                                            unsigned row_stride_elems) {
    unsigned long long ga = (unsigned long long)gptr;
    const unsigned td0 = 1u << 20, td1 = 1u << 20;   // generous tensor dims (no clip)
    v4u g0;
    g0[0] = 1u;                                           // count=1 (valid, user mode)
    g0[1] = lds_addr;                                     // lds byte address
    g0[2] = (unsigned)(ga & 0xFFFFFFFFull);               // global_addr[31:0]
    g0[3] = (unsigned)((ga >> 32) & 0x1FFFFFFull) | (2u << 30);  // addr[56:32] | type=2
    v8i g1;
    g1[0] = (int)(1u << 16);                              // data_size=1 -> 2 bytes
    g1[1] = (int)((td0 & 0xFFFFu) << 16);                 // atomic_barrier=0 | dim0 lo
    g1[2] = (int)((td0 >> 16) | ((td1 & 0xFFFFu) << 16)); // dim0 hi | dim1 lo
    g1[3] = (int)((td1 >> 16) | (tile_k << 16));          // dim1 hi | tile_dim0
    g1[4] = (int)tile_rows;                               // tile_dim1 | tile_dim2=0
    g1[5] = (int)row_stride_elems;                        // tensor_dim0_stride lo
    g1[6] = 0;
    g1[7] = 0;
    v4i z4 = {0, 0, 0, 0};
#if defined(__clang_major__) && (__clang_major__ >= 23)
    v8i z8 = {0, 0, 0, 0, 0, 0, 0, 0};
    __builtin_amdgcn_tensor_load_to_lds(g0, g1, z4, z4, z8, 0);
#else
    __builtin_amdgcn_tensor_load_to_lds(g0, g1, z4, z4, 0);
#endif
}

// ---------------------------------------------------------------------------
// fp32 -> bf16 convert (grid-stride)
// ---------------------------------------------------------------------------
__global__ __launch_bounds__(256) void cvt_f32_bf16(const float* __restrict__ in,
                                                    bf16* __restrict__ out, size_t n) {
    size_t i = (size_t)blockIdx.x * 256 + threadIdx.x;
    size_t stride = (size_t)gridDim.x * 256;
    for (; i < n; i += stride) out[i] = (bf16)in[i];
}

// ---------------------------------------------------------------------------
// QKV projection: Y = (X @ W^T + b) * scale, scattered into [B,H,S,Dh] bf16.
// Block tile 128x64, 8 waves each own 32x32 (4 WMMAs / K-step).
// A/B tiles staged via TDM tensor_load_to_lds, double-buffered: wave 0 issues
// the next tiles' DMA before computing, waits TENSORcnt just before barrier.
// ---------------------------------------------------------------------------
__global__ __launch_bounds__(256) void gemm_qkv_proj(const bf16* __restrict__ A,
                                                     const bf16* __restrict__ Bt,
                                                     const float* __restrict__ bias,
                                                     bf16* __restrict__ outH,
                                                     float scale) {
    __shared__ __align__(16) bf16 sA[2][128 * 32];
    __shared__ __align__(16) bf16 sB[2][64 * 32];
    const int K = D_MODEL;
    int m0 = blockIdx.x << 7, n0 = blockIdx.y << 6;
    int tid = threadIdx.x, lane = tid & 31, w = tid >> 5;
    int wr = (w & 3) << 5, wc = (w >> 2) << 5;
    v8f a00 = {}, a01 = {}, a10 = {}, a11 = {};

    if (w == 0) {
        tdm_load_2d(lds_off_u32(sA[0]), A  + (size_t)m0 * K, 32, 128, K);
        tdm_load_2d(lds_off_u32(sB[0]), Bt + (size_t)n0 * K, 32, 64, K);
        __builtin_amdgcn_s_wait_tensorcnt(0);
    }
    __syncthreads();

    for (int k0 = 0; k0 < K; k0 += 32) {
        int cur = (k0 >> 5) & 1;
        bool more = (k0 + 32) < K;
        if (w == 0 && more) {   // kick off DMA for next K-step into the other buffer
            tdm_load_2d(lds_off_u32(sA[cur ^ 1]), A  + (size_t)m0 * K + k0 + 32, 32, 128, K);
            tdm_load_2d(lds_off_u32(sB[cur ^ 1]), Bt + (size_t)n0 * K + k0 + 32, 32, 64, K);
        }
        v16bf fa0 = load_fragA(sA[cur], wr);
        v16bf fa1 = load_fragA(sA[cur], wr + 16);
        v16bf fb0 = load_fragB(sB[cur], wc);
        v16bf fb1 = load_fragB(sB[cur], wc + 16);
        a00 = WMMA_BF16(fa0, fb0, a00);
        a01 = WMMA_BF16(fa0, fb1, a01);
        a10 = WMMA_BF16(fa1, fb0, a10);
        a11 = WMMA_BF16(fa1, fb1, a11);
        if (w == 0 && more) __builtin_amdgcn_s_wait_tensorcnt(0);
        __syncthreads();
    }

    auto st = [&](const v8f& a, int mo, int no) {
        int mb = m0 + wr + mo + ((lane >> 4) << 3);
        int n  = n0 + wc + no + (lane & 15);
        int h = n >> 6, dh = n & 63;
        float bv = bias[n];
#pragma unroll
        for (int g = 0; g < 8; ++g) {
            int m = mb + g;
            int bi = m >> 11, s = m & (SQ - 1);
            outH[(((size_t)bi * N_HEAD + h) * SQ + s) * D_HEAD + dh] =
                (bf16)((a[g] + bv) * scale);
        }
    };
    st(a00, 0, 0); st(a01, 0, 16); st(a10, 16, 0); st(a11, 16, 16);
}

// ---------------------------------------------------------------------------
// Fused scores + softmax. One block owns a 64-row x 2048-col strip of one
// (b,h): computes logits = Qh Kh^T + mask via WMMA (Q strip cached in LDS the
// whole time), writes them, then softmaxes its own strip while it is L2-hot.
// ---------------------------------------------------------------------------
__global__ __launch_bounds__(256) void score_softmax(const bf16* __restrict__ Qh,
                                                     const bf16* __restrict__ Kh,
                                                     const float* __restrict__ mask,
                                                     float* __restrict__ score) {
    __shared__ __align__(16) bf16 sQ[2][64 * 32];
    __shared__ __align__(16) bf16 sK[2][64 * 32];
    int bh = blockIdx.z;
    int m0 = blockIdx.x << 6;
    const bf16* A  = Qh + (size_t)bh * SQ * D_HEAD;
    const bf16* Bt = Kh + (size_t)bh * SQ * D_HEAD;
    float* out = score + (size_t)bh * SQ * SQ;
    int tid = threadIdx.x, lane = tid & 31, w = tid >> 5;
    int wr = (w & 3) << 4, wc = (w >> 2) << 5;
    int r = tid >> 2, c = (tid & 3) << 3;

    // stage the whole 64x64 Q strip once (two K=32 halves)
    *(v8bf*)(&sQ[0][r * 32 + c]) = *(const v8bf*)(A + (size_t)(m0 + r) * D_HEAD + c);
    *(v8bf*)(&sQ[1][r * 32 + c]) = *(const v8bf*)(A + (size_t)(m0 + r) * D_HEAD + 32 + c);
    __syncthreads();

    for (int n0 = 0; n0 < SQ; n0 += 64) {
        *(v8bf*)(&sK[0][r * 32 + c]) = *(const v8bf*)(Bt + (size_t)(n0 + r) * D_HEAD + c);
        *(v8bf*)(&sK[1][r * 32 + c]) = *(const v8bf*)(Bt + (size_t)(n0 + r) * D_HEAD + 32 + c);
        __syncthreads();
        v8f acc0 = {}, acc1 = {};
#pragma unroll
        for (int kb = 0; kb < 2; ++kb) {
            v16bf a  = load_fragA(sQ[kb], wr);
            v16bf b0 = load_fragB(sK[kb], wc);
            v16bf b1 = load_fragB(sK[kb], wc + 16);
            acc0 = WMMA_BF16(a, b0, acc0);
            acc1 = WMMA_BF16(a, b1, acc1);
        }
        int mt = wr + ((lane >> 4) << 3);
        int na = n0 + wc + (lane & 15);
#pragma unroll
        for (int g = 0; g < 8; ++g) {
            int m = m0 + mt + g;
            out[(size_t)m * SQ + na]      = acc0[g] + mask[(size_t)m * SQ + na];
            out[(size_t)m * SQ + na + 16] = acc1[g] + mask[(size_t)m * SQ + na + 16];
        }
        __syncthreads();   // protect sK restage
    }

    __threadfence();       // make the strip's logits visible to all waves
    __syncthreads();

    // Softmax over own strip: 4 threads per row, 512 contiguous floats each.
    int row = tid >> 2, part = tid & 3;
    float4* p4 = (float4*)(out + (size_t)(m0 + row) * SQ + (part << 9));
    float lmax = -3.0e38f;
    for (int i = 0; i < 128; ++i) {
        float4 f = p4[i];
        lmax = fmaxf(lmax, fmaxf(fmaxf(f.x, f.y), fmaxf(f.z, f.w)));
    }
    lmax = fmaxf(lmax, __shfl_xor(lmax, 1, 32));
    lmax = fmaxf(lmax, __shfl_xor(lmax, 2, 32));
    float lsum = 0.0f;
    for (int i = 0; i < 128; ++i) {
        float4 f = p4[i];
        f.x = __expf(f.x - lmax); f.y = __expf(f.y - lmax);
        f.z = __expf(f.z - lmax); f.w = __expf(f.w - lmax);
        lsum += (f.x + f.y) + (f.z + f.w);
        p4[i] = f;
    }
    lsum += __shfl_xor(lsum, 1, 32);
    lsum += __shfl_xor(lsum, 2, 32);
    float inv = 1.0f / lsum;
    for (int i = 0; i < 128; ++i) {
        float4 f = p4[i];
        f.x *= inv; f.y *= inv; f.z *= inv; f.w *= inv;
        p4[i] = f;
    }
}

// ---------------------------------------------------------------------------
// P @ V: A = probs (fp32 -> bf16 during staging), Bt[n=dh][k] = Vh[k][dh]
// (transposed while staging). Block tile 128x64, 4 WMMAs/wave/K-step.
// Output bf16 into concat-head layout Ax[B, S, D_MODEL].
// ---------------------------------------------------------------------------
__global__ __launch_bounds__(256) void gemm_av(const float* __restrict__ score,
                                               const bf16* __restrict__ Vh,
                                               bf16* __restrict__ Ax) {
    __shared__ __align__(16) bf16 sA[128 * 32];
    __shared__ __align__(16) bf16 sB[64 * 32];
    int bh = blockIdx.z, bi = bh >> 4, h = bh & 15;
    const float* P = score + (size_t)bh * SQ * SQ;
    const bf16*  V = Vh    + (size_t)bh * SQ * D_HEAD;
    int m0 = blockIdx.x << 7;
    int tid = threadIdx.x, lane = tid & 31, w = tid >> 5;
    int wr = (w & 3) << 5, wc = (w >> 2) << 5;
    int r = tid >> 1, c = (tid & 1) << 4;   // 16 fp32 per thread for the A tile
    int kk = tid & 31;
    v8f a00 = {}, a01 = {}, a10 = {}, a11 = {};
    for (int k0 = 0; k0 < SQ; k0 += 32) {
        const float4* s4 = (const float4*)(P + (size_t)(m0 + r) * SQ + k0 + c);
#pragma unroll
        for (int j = 0; j < 4; ++j) {
            float4 f = s4[j];
            bf16* d = &sA[r * 32 + c + (j << 2)];
            d[0] = (bf16)f.x; d[1] = (bf16)f.y; d[2] = (bf16)f.z; d[3] = (bf16)f.w;
        }
#pragma unroll
        for (int j = tid >> 5; j < 64; j += 8)
            sB[j * 32 + kk] = V[(size_t)(k0 + kk) * D_HEAD + j];
        __syncthreads();
        v16bf fa0 = load_fragA(sA, wr);
        v16bf fa1 = load_fragA(sA, wr + 16);
        v16bf fb0 = load_fragB(sB, wc);
        v16bf fb1 = load_fragB(sB, wc + 16);
        a00 = WMMA_BF16(fa0, fb0, a00);
        a01 = WMMA_BF16(fa0, fb1, a01);
        a10 = WMMA_BF16(fa1, fb0, a10);
        a11 = WMMA_BF16(fa1, fb1, a11);
        __syncthreads();
    }
    auto st = [&](const v8f& a, int mo, int no) {
        int mb = m0 + wr + mo + ((lane >> 4) << 3);
        int n  = wc + no + (lane & 15);
#pragma unroll
        for (int g = 0; g < 8; ++g) {
            int m = mb + g;
            Ax[((size_t)bi * SQ + m) * D_MODEL + h * D_HEAD + n] = (bf16)a[g];
        }
    };
    st(a00, 0, 0); st(a01, 0, 16); st(a10, 16, 0); st(a11, 16, 16);
}

// ---------------------------------------------------------------------------
// Output projection: X = Ax @ Wo^T + b_o, fp32 row-major into d_out.
// Same TDM double-buffered structure as the QKV projections.
// ---------------------------------------------------------------------------
__global__ __launch_bounds__(256) void gemm_out_proj(const bf16* __restrict__ A,
                                                     const bf16* __restrict__ Bt,
                                                     const float* __restrict__ bias,
                                                     float* __restrict__ out) {
    __shared__ __align__(16) bf16 sA[2][128 * 32];
    __shared__ __align__(16) bf16 sB[2][64 * 32];
    const int K = D_MODEL;
    int m0 = blockIdx.x << 7, n0 = blockIdx.y << 6;
    int tid = threadIdx.x, lane = tid & 31, w = tid >> 5;
    int wr = (w & 3) << 5, wc = (w >> 2) << 5;
    v8f a00 = {}, a01 = {}, a10 = {}, a11 = {};

    if (w == 0) {
        tdm_load_2d(lds_off_u32(sA[0]), A  + (size_t)m0 * K, 32, 128, K);
        tdm_load_2d(lds_off_u32(sB[0]), Bt + (size_t)n0 * K, 32, 64, K);
        __builtin_amdgcn_s_wait_tensorcnt(0);
    }
    __syncthreads();

    for (int k0 = 0; k0 < K; k0 += 32) {
        int cur = (k0 >> 5) & 1;
        bool more = (k0 + 32) < K;
        if (w == 0 && more) {
            tdm_load_2d(lds_off_u32(sA[cur ^ 1]), A  + (size_t)m0 * K + k0 + 32, 32, 128, K);
            tdm_load_2d(lds_off_u32(sB[cur ^ 1]), Bt + (size_t)n0 * K + k0 + 32, 32, 64, K);
        }
        v16bf fa0 = load_fragA(sA[cur], wr);
        v16bf fa1 = load_fragA(sA[cur], wr + 16);
        v16bf fb0 = load_fragB(sB[cur], wc);
        v16bf fb1 = load_fragB(sB[cur], wc + 16);
        a00 = WMMA_BF16(fa0, fb0, a00);
        a01 = WMMA_BF16(fa0, fb1, a01);
        a10 = WMMA_BF16(fa1, fb0, a10);
        a11 = WMMA_BF16(fa1, fb1, a11);
        if (w == 0 && more) __builtin_amdgcn_s_wait_tensorcnt(0);
        __syncthreads();
    }

    auto st = [&](const v8f& a, int mo, int no) {
        int mb = m0 + wr + mo + ((lane >> 4) << 3);
        int n  = n0 + wc + no + (lane & 15);
        float bv = bias[n];
#pragma unroll
        for (int g = 0; g < 8; ++g) {
            int m = mb + g;
            out[(size_t)m * D_MODEL + n] = a[g] + bv;
        }
    };
    st(a00, 0, 0); st(a01, 0, 16); st(a10, 16, 0); st(a11, 16, 16);
}

// ---------------------------------------------------------------------------
// Host-side orchestration
// ---------------------------------------------------------------------------
extern "C" void kernel_launch(void* const* d_in, const int* in_sizes, int n_in,
                              void* d_out, int out_size, void* d_ws, size_t ws_size,
                              hipStream_t stream) {
    const float* q    = (const float*)d_in[0];
    const float* k    = (const float*)d_in[1];
    const float* v    = (const float*)d_in[2];
    const float* mask = (const float*)d_in[3];
    const float* w_q  = (const float*)d_in[4];
    const float* b_q  = (const float*)d_in[5];
    const float* w_k  = (const float*)d_in[6];
    const float* b_k  = (const float*)d_in[7];
    const float* w_v  = (const float*)d_in[8];
    const float* b_v  = (const float*)d_in[9];
    const float* w_o  = (const float*)d_in[10];
    const float* b_o  = (const float*)d_in[11];

    float* out_x     = (float*)d_out;                               // [B,S,D]
    float* out_score = out_x + (size_t)NB * SQ * D_MODEL;           // [B,H,S,S]

    char* ws = (char*)d_ws;
    size_t off = 0;
    auto take = [&](size_t bytes) -> void* {
        void* p = ws + off;
        off += (bytes + 255) & ~(size_t)255;
        return p;
    };
    const size_t xe = (size_t)MTOT * D_MODEL;
    const size_t we = (size_t)D_MODEL * D_MODEL;
    bf16* qb  = (bf16*)take(xe * 2);
    bf16* kb  = (bf16*)take(xe * 2);
    bf16* vb  = (bf16*)take(xe * 2);
    bf16* wqb = (bf16*)take(we * 2);
    bf16* wkb = (bf16*)take(we * 2);
    bf16* wvb = (bf16*)take(we * 2);
    bf16* wob = (bf16*)take(we * 2);
    bf16* Qh  = (bf16*)take(xe * 2);
    bf16* Kh  = (bf16*)take(xe * 2);
    bf16* Vh  = (bf16*)take(xe * 2);
    bf16* Ax  = (bf16*)take(xe * 2);

    // 1) convert fp32 -> bf16
    cvt_f32_bf16<<<4096, 256, 0, stream>>>(q,   qb,  xe);
    cvt_f32_bf16<<<4096, 256, 0, stream>>>(k,   kb,  xe);
    cvt_f32_bf16<<<4096, 256, 0, stream>>>(v,   vb,  xe);
    cvt_f32_bf16<<<2048, 256, 0, stream>>>(w_q, wqb, we);
    cvt_f32_bf16<<<2048, 256, 0, stream>>>(w_k, wkb, we);
    cvt_f32_bf16<<<2048, 256, 0, stream>>>(w_v, wvb, we);
    cvt_f32_bf16<<<2048, 256, 0, stream>>>(w_o, wob, we);

    // 2) projections (Q pre-scaled by 1/sqrt(Dh) = 0.125)
    dim3 pg(MTOT / 128, D_MODEL / 64);
    gemm_qkv_proj<<<pg, 256, 0, stream>>>(qb, wqb, b_q, Qh, 0.125f);
    gemm_qkv_proj<<<pg, 256, 0, stream>>>(kb, wkb, b_k, Kh, 1.0f);
    gemm_qkv_proj<<<pg, 256, 0, stream>>>(vb, wvb, b_v, Vh, 1.0f);

    // 3) fused logits + mask + softmax (the probs ARE the second output)
    score_softmax<<<dim3(SQ / 64, 1, NB * N_HEAD), 256, 0, stream>>>(Qh, Kh, mask, out_score);

    // 4) P @ V -> concat-head bf16
    gemm_av<<<dim3(SQ / 128, 1, NB * N_HEAD), 256, 0, stream>>>(out_score, Vh, Ax);

    // 5) output projection -> fp32 d_out
    gemm_out_proj<<<pg, 256, 0, stream>>>(Ax, wob, b_o, out_x);
}
